// AutoIntTPPSameInfluence_3410204033605
// MI455X (gfx1250) — matrix operand order repaired and checked by reading, not compiled, not run on hardware
//
#include <hip/hip_runtime.h>
#include <hip/hip_bf16.h>

typedef __attribute__((ext_vector_type(16))) _Float16 v16h;
typedef __attribute__((ext_vector_type(8)))  _Float16 v8h;
typedef __attribute__((ext_vector_type(8)))  float    v8f;

#define BATCH 16
#define LMAX  320
#define HID   64
#define NW    8          // waves per block (256 threads, wave32)
#define ROWBLOCKS 40     // ceil(319 rows / 8 waves)
#define INTBLOCKS 3      // 3*8 waves >= 20 integral groups
#define T_END_F 100.0f

__device__ __forceinline__ void lds_fence() {
  __builtin_amdgcn_wave_barrier();
  asm volatile("s_wait_dscnt 0" ::: "memory");
}

// Hardware tanh on CDNA5 (V_TANH_F32); branch-free exp2-based fallback.
__device__ __forceinline__ float fast_tanh(float x) {
#if __has_builtin(__builtin_amdgcn_tanhf)
  return __builtin_amdgcn_tanhf(x);
#elif __has_builtin(__builtin_amdgcn_tanh_f32)
  return __builtin_amdgcn_tanh_f32(x);
#else
  // tanh(x) = 1 - 2/(exp2(2*log2(e)*x) + 1); saturates correctly at +/-inf.
  const float e = __builtin_amdgcn_exp2f(x * 2.88539008177793f);
  return 1.0f - 2.0f * __builtin_amdgcn_rcpf(e + 1.0f);
#endif
}

__device__ __forceinline__ v8f wmma16(v16h a, v16h b, v8f c) {
  // D = A(16x32 f16) x B(32x16 f16) + C(16x16 f32)
  return __builtin_amdgcn_wmma_f32_16x16x32_f16(false, a, false, b, (short)0, c,
                                                false, false);
}

// Gather a 16x32 f16 A-fragment from an LDS tile stored as [16 rows][64 cols].
// Per ISA 16-bit A layout: lane<16 -> M=lane, halves map to K = {c..c+7, c+16..c+23},
// c = 0 for lanes 0-15, c = 8 for lanes 16-31.  Two contiguous b128 loads.
__device__ __forceinline__ v16h load_a_frag(const _Float16* base, int lane, int kc) {
  const int M = lane & 15;
  const int c = (lane & 16) ? 8 : 0;
  const _Float16* p = base + M * 64 + kc * 32 + c;
  v8h lo = *(const v8h*)(p);
  v8h hi = *(const v8h*)(p + 16);
  v16h r;
#pragma unroll
  for (int q = 0; q < 8; ++q) { r[q] = lo[q]; r[q + 8] = hi[q]; }
  return r;
}

// Run the MLP (and optionally its input-derivative via the JVP chain) on 16
// scalar samples in sX_w[0..15].  On exit vout[r] holds this lane's partial of
// layer-4 (sum over its 4 output units of W4[o]*g3 or W4[o]*h3).  Reducing over
// lanes (xor 1,2,4,8) yields per-sample results: lane 0 -> samples 0..7,
// lane 16 -> samples 8..15.
template <bool DERIV>
__device__ __forceinline__ void mlp_group(
    int lane, const float* sX_w, _Float16* hbuf, _Float16* gbuf,
    const float* sW1, const float* sB1,
    const _Float16* sFragW2, const float* sB2,
    const _Float16* sFragW3, const float* sB3,
    const float* sW4, float vout[8]) {
  // ---- layer 1: scalar -> 64, elementwise; packed b128 stores ----
  {
    const int s  = lane & 15;
    const int k0 = (lane & 16) ? 32 : 0;
    const float x = sX_w[s];
    _Float16 hv[32], gv[32];
#pragma unroll
    for (int kk = 0; kk < 32; ++kk) {
      const int k = k0 + kk;
      const float wv = sW1[k];
      const float h  = fast_tanh(fmaf(wv, x, sB1[k]));
      hv[kk] = (_Float16)h;
      if (DERIV) gv[kk] = (_Float16)(wv * (1.0f - h * h));
    }
    _Float16* hp = hbuf + s * 64 + k0;
    _Float16* gp = gbuf + s * 64 + k0;
#pragma unroll
    for (int q = 0; q < 4; ++q) {
      v8h ph, pg;
#pragma unroll
      for (int e = 0; e < 8; ++e) { ph[e] = hv[8 * q + e]; if (DERIV) pg[e] = gv[8 * q + e]; }
      *(v8h*)(hp + 8 * q) = ph;
      if (DERIV) *(v8h*)(gp + 8 * q) = pg;
    }
  }
  lds_fence();

  const int n  = lane & 15;
  const int sh = (lane & 16) ? 8 : 0;

  // ---- layer 2: [16x64] @ W2^T via WMMA ----
  v8f accH[4], accG[4];
#pragma unroll
  for (int nt = 0; nt < 4; ++nt) { accH[nt] = {}; if (DERIV) accG[nt] = {}; }
#pragma unroll
  for (int kc = 0; kc < 2; ++kc) {
    v16h aH = load_a_frag(hbuf, lane, kc);
    v16h aG{};
    if (DERIV) aG = load_a_frag(gbuf, lane, kc);
#pragma unroll
    for (int nt = 0; nt < 4; ++nt) {
      v16h bf = *(const v16h*)&sFragW2[((kc * 4 + nt) * 32 + lane) * 16];
      accH[nt] = wmma16(aH, bf, accH[nt]);
      if (DERIV) accG[nt] = wmma16(aG, bf, accG[nt]);
    }
  }
  // bias + tanh (+ tangent) back to LDS tiles
#pragma unroll
  for (int nt = 0; nt < 4; ++nt) {
    const int o = nt * 16 + n;
    const float bo = sB2[o];
#pragma unroll
    for (int r = 0; r < 8; ++r) {
      const float h = fast_tanh(accH[nt][r] + bo);
      hbuf[(sh + r) * 64 + o] = (_Float16)h;
      if (DERIV) gbuf[(sh + r) * 64 + o] = (_Float16)((1.0f - h * h) * accG[nt][r]);
    }
  }
  lds_fence();

  // ---- layer 3 ----
#pragma unroll
  for (int nt = 0; nt < 4; ++nt) { accH[nt] = {}; if (DERIV) accG[nt] = {}; }
#pragma unroll
  for (int kc = 0; kc < 2; ++kc) {
    v16h aH = load_a_frag(hbuf, lane, kc);
    v16h aG{};
    if (DERIV) aG = load_a_frag(gbuf, lane, kc);
#pragma unroll
    for (int nt = 0; nt < 4; ++nt) {
      v16h bf = *(const v16h*)&sFragW3[((kc * 4 + nt) * 32 + lane) * 16];
      accH[nt] = wmma16(aH, bf, accH[nt]);
      if (DERIV) accG[nt] = wmma16(aG, bf, accG[nt]);
    }
  }

  // ---- layer 4 partials (bias b4 cancels in all uses) ----
#pragma unroll
  for (int r = 0; r < 8; ++r) vout[r] = 0.0f;
#pragma unroll
  for (int nt = 0; nt < 4; ++nt) {
    const int o = nt * 16 + n;
    const float w4 = sW4[o];
    const float bo = sB3[o];
#pragma unroll
    for (int r = 0; r < 8; ++r) {
      const float h = fast_tanh(accH[nt][r] + bo);
      if (DERIV) vout[r] += w4 * (1.0f - h * h) * accG[nt][r];
      else       vout[r] += w4 * h;
    }
  }
}

__global__ void __launch_bounds__(256) tpp_init_ws(float* ws) {
  if (threadIdx.x < 2) ws[threadIdx.x] = 0.0f;
}

__global__ void __launch_bounds__(256) tpp_main_kernel(
    const float* __restrict__ seq_pads, const float* __restrict__ bg,
    const float* __restrict__ W1, const float* __restrict__ b1,
    const float* __restrict__ W2, const float* __restrict__ b2,
    const float* __restrict__ W3, const float* __restrict__ b3,
    const float* __restrict__ W4, const int* __restrict__ seq_lens,
    float* __restrict__ ws) {
  __shared__ _Float16 sFragW2[4096] __attribute__((aligned(32)));
  __shared__ _Float16 sFragW3[4096] __attribute__((aligned(32)));
  __shared__ float sW1[HID], sB1[HID], sB2[HID], sB3[HID], sW4v[HID];
  __shared__ float sT[LMAX];
  __shared__ _Float16 sH[NW][16 * 64] __attribute__((aligned(32)));
  __shared__ _Float16 sG[NW][16 * 64] __attribute__((aligned(32)));
  __shared__ float sX[NW][16];

  const int tid  = threadIdx.x;
  const int b    = blockIdx.y;
  const int lane = tid & 31;
  const int w    = tid >> 5;

  // ---- block init: stage t row, biases, and WMMA B-fragments of W2/W3 ----
  for (int k = tid; k < LMAX; k += 256) sT[k] = seq_pads[b * LMAX + k];
  if (tid < HID) {
    sW1[tid] = W1[tid]; sB1[tid] = b1[tid];
    sB2[tid] = b2[tid]; sB3[tid] = b3[tid];
    sW4v[tid] = W4[tid];
  }
  {
    // fragment element: B[in][o] = W[o][in]; lane n = lane&15 -> N,
    // elements map to K = {c..c+7, c+16..c+23}, c = 0 for lanes 0-15, c = 8 for lanes 16-31.
    const int kc = tid >> 7;
    const int nt = (tid >> 5) & 3;
    const int ln = tid & 31;
    const int o   = nt * 16 + (ln & 15);
    const int in0 = kc * 32 + ((ln & 16) ? 8 : 0);
#pragma unroll
    for (int h = 0; h < 16; ++h) {
      sFragW2[tid * 16 + h] = (_Float16)W2[o * 64 + in0 + h + (h >= 8 ? 8 : 0)];
      sFragW3[tid * 16 + h] = (_Float16)W3[o * 64 + in0 + h + (h >= 8 ? 8 : 0)];
    }
  }
  __syncthreads();

  const int bx = blockIdx.x;
  const float bg0 = bg[0];

  if (bx < ROWBLOCKS) {
    // ======== pairwise dF rows: wave handles row i (target event i+1) ========
    const int i  = bx * NW + w;
    const int sl = seq_lens[b];
    if (i <= LMAX - 2 && (i + 1) < sl) {
      const float ti1 = sT[i + 1];
      const int ngroups = (i + 16) >> 4;   // ceil((i+1)/16)
      float acc = 0.0f;
      for (int jg = 0; jg < ngroups; ++jg) {
        if (lane < 16) {
          const int j = jg * 16 + lane;
          sX[w][lane] = (j <= i) ? (ti1 - sT[j]) : 0.0f;
        }
        lds_fence();
        float v[8];
        mlp_group<true>(lane, sX[w], sH[w], sG[w], sW1, sB1,
                        sFragW2, sB2, sFragW3, sB3, sW4v, v);
#pragma unroll
        for (int r = 0; r < 8; ++r) {
          float tv = v[r];
          tv += __shfl_xor(tv, 1);
          tv += __shfl_xor(tv, 2);
          tv += __shfl_xor(tv, 4);
          tv += __shfl_xor(tv, 8);
          v[r] = tv;
        }
        if ((lane & 15) == 0) {
          const int base = jg * 16 + ((lane >> 4) * 8);
#pragma unroll
          for (int r = 0; r < 8; ++r)
            if (base + r <= i) acc += v[r];
        }
      }
      acc += __shfl_xor(acc, 16);
      if (lane == 0) atomicAdd(&ws[0], logf(acc + bg0));
    }
  } else {
    // ======== integral term: F(T_end - t_k); masked lanes evaluate F(0) ========
    const int g = (bx - ROWBLOCKS) * NW + w;
    if (g < LMAX / 16) {
      const int sl = seq_lens[b];
      if (lane < 16) {
        const int k = g * 16 + lane;
        sX[w][lane] = (k < sl) ? (T_END_F - sT[k]) : 0.0f;
      }
      lds_fence();
      float v[8];
      mlp_group<false>(lane, sX[w], sH[w], sG[w], sW1, sB1,
                       sFragW2, sB2, sFragW3, sB3, sW4v, v);
#pragma unroll
      for (int r = 0; r < 8; ++r) {
        float tv = v[r];
        tv += __shfl_xor(tv, 1);
        tv += __shfl_xor(tv, 2);
        tv += __shfl_xor(tv, 4);
        tv += __shfl_xor(tv, 8);
        v[r] = tv;
      }
      float part = 0.0f;
      if ((lane & 15) == 0) {
#pragma unroll
        for (int r = 0; r < 8; ++r) part += v[r];
      }
      part += __shfl_xor(part, 16);
      if (lane == 0) atomicAdd(&ws[1], part);   // sum of F~(x) (no b4) over 16 samples
    }
  }
}

__global__ void __launch_bounds__(64) tpp_finalize(
    const float* __restrict__ bg,
    const float* __restrict__ b1, const float* __restrict__ W2,
    const float* __restrict__ b2, const float* __restrict__ W3,
    const float* __restrict__ b3, const float* __restrict__ W4,
    const float* __restrict__ ws, float* __restrict__ out) {
  if (threadIdx.x != 0 || blockIdx.x != 0) return;
  // F~(0) (without b4; b4 cancels in F(x)-F(0))
  float h[HID], h2[HID];
  for (int k = 0; k < HID; ++k) h[k] = fast_tanh(b1[k]);        // x = 0
  for (int o = 0; o < HID; ++o) {
    float a = b2[o];
    for (int in = 0; in < HID; ++in) a = fmaf(W2[o * HID + in], h[in], a);
    h2[o] = fast_tanh(a);
  }
  for (int o = 0; o < HID; ++o) {
    float a = b3[o];
    for (int in = 0; in < HID; ++in) a = fmaf(W3[o * HID + in], h2[in], a);
    h[o] = fast_tanh(a);
  }
  float F0 = 0.0f;
  for (int o = 0; o < HID; ++o) F0 = fmaf(W4[o], h[o], F0);

  const float bg0 = bg[0];
  const float sum_log = ws[0] + (float)BATCH * logf(bg0);       // + log(lambs[b,0])
  const float sum_int = ws[1] - (float)BATCH * (float)LMAX * F0 +
                        (float)BATCH * T_END_F * bg0;
  out[0] = -(sum_log - sum_int) / (float)BATCH;
}

extern "C" void kernel_launch(void* const* d_in, const int* in_sizes, int n_in,
                              void* d_out, int out_size, void* d_ws, size_t ws_size,
                              hipStream_t stream) {
  const float* seq_pads   = (const float*)d_in[0];
  const float* background = (const float*)d_in[1];
  const float* W1         = (const float*)d_in[2];
  const float* b1         = (const float*)d_in[3];
  const float* W2         = (const float*)d_in[4];
  const float* b2         = (const float*)d_in[5];
  const float* W3         = (const float*)d_in[6];
  const float* b3         = (const float*)d_in[7];
  const float* W4         = (const float*)d_in[8];
  // d_in[9] = b4 (cancels analytically), d_in[10] = seq_lens
  const int*   seq_lens   = (const int*)d_in[10];
  float* ws  = (float*)d_ws;
  float* out = (float*)d_out;

  tpp_init_ws<<<1, 32, 0, stream>>>(ws);
  tpp_main_kernel<<<dim3(ROWBLOCKS + INTBLOCKS, BATCH), 256, 0, stream>>>(
      seq_pads, background, W1, b1, W2, b2, W3, b3, W4, seq_lens, ws);
  tpp_finalize<<<1, 64, 0, stream>>>(background, b1, W2, b2, W3, b3, W4, ws, out);
}
